// TaylorAttention_51917564674053
// MI455X (gfx1250) — matrix operand order repaired and checked by reading, not compiled, not run on hardware
//
#include <hip/hip_runtime.h>
#include <hip/hip_bf16.h>
#include <stdint.h>

// ---------------------------------------------------------------------------
// Taylor attention, MI455X (gfx1250, wave32, WMMA 16x16x32 bf16 + TDM)
//   B=8, N=1024, C=768, H=12, Dh=64, ORDER=2
// Pipeline:
//   1) cvt x -> bf16            (Xb)
//   2) pack W_qkv, W_proj into WMMA-B-fragment-linear bf16 layout
//   3) QKV GEMM (bf16 WMMA, f32 acc) -> Qb(scaled)/Kb/Vb  [B,H,N,Dh] bf16
//   4) streaming attention: K tile staged by Tensor Data Mover (TDM) with
//      hardware LDS row padding; S=QK^T (WMMA), Taylor+relu, rowsum,
//      O += T*V (WMMA via LDS round-trip), normalize -> Ob [B,N,C] bf16
//   5) proj GEMM (bf16 WMMA) + bias -> f32 out
// ---------------------------------------------------------------------------

typedef __bf16 bf16_t;
typedef __attribute__((ext_vector_type(16))) __bf16 v16bf;
typedef __attribute__((ext_vector_type(8)))  __bf16 v8bf;
typedef __attribute__((ext_vector_type(8)))  float  v8f;
typedef __attribute__((ext_vector_type(4)))  unsigned int u32x4;
typedef __attribute__((ext_vector_type(8)))  int i32x8;
typedef __attribute__((ext_vector_type(4)))  int i32x4;

#if __has_builtin(__builtin_amdgcn_tensor_load_to_lds) && \
    __has_builtin(__builtin_amdgcn_s_wait_tensorcnt)
#define ATTN_TDM 1
#else
#define ATTN_TDM 0
#endif

#define DEVINL __device__ __forceinline__

DEVINL bf16_t f2bf(float f){
  unsigned u = __builtin_bit_cast(unsigned, f);
  u += 0x7FFFu + ((u >> 16) & 1u);            // round-to-nearest-even
  unsigned short h = (unsigned short)(u >> 16);
  return __builtin_bit_cast(bf16_t, h);
}

DEVINL v8f zero8(){ v8f z = {0.f,0.f,0.f,0.f,0.f,0.f,0.f,0.f}; return z; }

DEVINL v8f wmma_bf16(v16bf a, v16bf b, v8f c){
  // D = A(16x32) * B(32x16) + C, f32 accumulate
  return __builtin_amdgcn_wmma_f32_16x16x32_bf16(
      /*neg_a=*/false, a, /*neg_b=*/false, b,
      /*c_mod=*/(short)0, c, /*reuse_a=*/false, /*reuse_b=*/false);
}

// A fragment (16x32 bf16) from a row-major source (row stride in elements).
// Lane L<16 : row=L, K 0..15 contiguous; lane>=16: K 16..31 contiguous.
DEVINL v16bf load_a_frag(const bf16_t* base, int stride, int lane){
  const int m  = lane & 15;
  const int hi = lane >> 4;
  const bf16_t* p = base + (size_t)m * stride + hi * 16;
  v8bf lo = *(const v8bf*)(p);
  v8bf hh = *(const v8bf*)(p + 8);
  v16bf a;
#pragma unroll
  for (int i = 0; i < 8; ++i){ a[i] = lo[i]; a[8+i] = hh[i]; }
  return a;
}

// B fragment (32x16 bf16) where WMMA "N" selects a row (stride apart) and the
// contraction index runs contiguously within the row.
// Lane L: N = L&15; K = (L<16 ? 0..15 : 16..31) contiguous.
DEVINL v16bf load_b_frag(const bf16_t* base, int stride, int lane){
  const int n  = lane & 15;
  const int hi = lane >> 4;
  const bf16_t* p = base + (size_t)n * stride + hi * 16;
  v8bf lo = *(const v8bf*)(p);
  v8bf hh = *(const v8bf*)(p + 8);
  v16bf b;
#pragma unroll
  for (int i = 0; i < 8; ++i){ b[i] = lo[i]; b[8+i] = hh[i]; }
  return b;
}

#if ATTN_TDM
// Issue a TDM load of a 64x64 bf16 tile (rows = tile_dim1, row len = 64 elems)
// out of a [1024 x 64] bf16 tensor into LDS at lds_byte_addr, padding each
// 32-dword (128B) row with 4 dwords (16B) -> LDS row stride 144B (72 elems).
DEVINL void tdm_load_tile_64x64(uint64_t gaddr, unsigned int lds_byte_addr){
  u32x4 g0;
  g0[0] = 1u;                                   // count=1, user mode, no gather
  g0[1] = lds_byte_addr;                        // D# bits 63:32 lds_addr
  g0[2] = (unsigned int)(gaddr & 0xFFFFFFFFull);          // global_addr[31:0]
  g0[3] = (unsigned int)((gaddr >> 32) & 0x01FFFFFFull)   // global_addr[56:32]
        | 0x80000000u;                                    // type = 2 ("image")
  i32x8 g1;
  g1[0] = (int)((1u << 16)        // data_size = 1 -> 2 bytes
              | (1u << 20)        // pad_enable
              | (4u << 22)        // pad_interval: 2^(4+1) = 32 dwords
              | (3u << 25));      // pad_amount : 3+1 = 4 dwords (16B)
  g1[1] = (int)(64u << 16);       // tensor_dim0 = 64 (bits 79:48, low half)
  g1[2] = (int)(1024u << 16);     // tensor_dim0 hi=0 | tensor_dim1 = 1024 (lo)
  g1[3] = (int)(64u << 16);       // tensor_dim1 hi=0 | tile_dim0 = 64
  g1[4] = 64;                     // tile_dim1 = 64, tile_dim2 = 0
  g1[5] = 64;                     // tensor_dim0_stride = 64 (elements)
  g1[6] = 0;                      // stride hi | tensor_dim1_stride lo = 0
  g1[7] = 0;
  i32x4 z4 = {0, 0, 0, 0};        // groups 2/3 unused (2-D tensor)
  i32x8 z8 = {0, 0, 0, 0, 0, 0, 0, 0};
  // clang-23 / therock form: 6 args (g0, g1, g2, g3, extra, cpol)
  __builtin_amdgcn_tensor_load_to_lds(g0, g1, z4, z4, z8, 0);
}
#endif

// ---------------------------------------------------------------------------
// 1) f32 -> bf16 conversion of x (8 elems / thread)
__global__ void k_cvt_x(const float* __restrict__ x, bf16_t* __restrict__ Xb, int n){
  int i0 = (blockIdx.x * blockDim.x + threadIdx.x) * 8;
  if (i0 >= n) return;
  float4 f0 = *(const float4*)(x + i0);
  float4 f1 = *(const float4*)(x + i0 + 4);
  v8bf o;
  o[0]=f2bf(f0.x); o[1]=f2bf(f0.y); o[2]=f2bf(f0.z); o[3]=f2bf(f0.w);
  o[4]=f2bf(f1.x); o[5]=f2bf(f1.y); o[6]=f2bf(f1.z); o[7]=f2bf(f1.w);
  *(v8bf*)(Xb + i0) = o;
}

// ---------------------------------------------------------------------------
// 2) pack f32 weight [Kdim x Ncols] into B-fragment-linear bf16:
//    tile(kt,nt) = 32x16; 32 lanes x 16 contiguous K-values each.
__global__ void k_pack_w(const float* __restrict__ W, bf16_t* __restrict__ Wp,
                         int Kdim, int Ncols){
  int idx = blockIdx.x * blockDim.x + threadIdx.x;
  int ntn = Ncols / 16;
  int total = (Kdim / 32) * ntn * 32;
  if (idx >= total) return;
  int tile = idx >> 5, lane = idx & 31;
  int kt = tile / ntn, nt = tile % ntn;
  int n  = nt * 16 + (lane & 15);
  int k0 = kt * 32 + (lane >> 4) * 16;
  bf16_t* dst = Wp + (size_t)tile * 512 + (size_t)lane * 16;
#pragma unroll
  for (int i = 0; i < 16; ++i)
    dst[i] = f2bf(W[(size_t)(k0 + i) * Ncols + n]);
}

// ---------------------------------------------------------------------------
// 3) QKV GEMM: Xb[8192x768] @ Wqkv[768x2304] + bias, scatter into
//    Qb (x0.125) / Kb / Vb as [B,H,N,Dh] bf16. Wave = 32x64 output tile.
__global__ void __launch_bounds__(128) k_qkv_gemm(
    const bf16_t* __restrict__ Xb, const bf16_t* __restrict__ Wp,
    const float* __restrict__ bias,
    bf16_t* __restrict__ Qb, bf16_t* __restrict__ Kb, bf16_t* __restrict__ Vb)
{
  const int Cdim = 768, NC = 2304, Hh = 12, Dh = 64, Npos = 1024;
  const int lane = threadIdx.x & 31, wave = threadIdx.x >> 5;
  const int m0 = (blockIdx.x * 4 + wave) * 32;
  const int n0 = blockIdx.y * 64;
  const int nt16 = n0 >> 4;

  v8f acc[2][4];
#pragma unroll
  for (int g = 0; g < 2; ++g)
#pragma unroll
    for (int j = 0; j < 4; ++j) acc[g][j] = zero8();

  for (int kk = 0; kk < Cdim; kk += 32){
    v16bf a0 = load_a_frag(Xb + (size_t)m0 * Cdim + kk,        Cdim, lane);
    v16bf a1 = load_a_frag(Xb + (size_t)(m0 + 16) * Cdim + kk, Cdim, lane);
    const int kt = kk >> 5;
    const bf16_t* wbase = Wp + ((size_t)kt * (NC/16) + nt16) * 512 + (size_t)lane * 16;
    if (kk + 32 < Cdim)
      __builtin_prefetch(wbase + (size_t)(NC/16) * 512, 0, 1); // next K-slab
#pragma unroll
    for (int j = 0; j < 4; ++j){
      v16bf b = *(const v16bf*)(wbase + (size_t)j * 512);
      acc[0][j] = wmma_bf16(a0, b, acc[0][j]);
      acc[1][j] = wmma_bf16(a1, b, acc[1][j]);
    }
  }

  const int hi = lane >> 4;
#pragma unroll
  for (int g = 0; g < 2; ++g)
#pragma unroll
    for (int j = 0; j < 4; ++j)
#pragma unroll
      for (int r = 0; r < 8; ++r){
        int row = m0 + g * 16 + r + hi * 8;
        int col = n0 + j * 16 + (lane & 15);
        float v = acc[g][j][r] + bias[col];
        int bb = row >> 10, n = row & 1023;
        int which = col / 768;
        int rem = col - which * 768;
        int hh = rem >> 6, d = rem & 63;
        size_t off = (((size_t)bb * Hh + hh) * (size_t)Npos + n) * Dh + d;
        if (which == 0)      Qb[off] = f2bf(v * 0.125f);   // fold Dh^-0.5
        else if (which == 1) Kb[off] = f2bf(v);
        else                 Vb[off] = f2bf(v);
      }
}

// ---------------------------------------------------------------------------
// 4) streaming Taylor attention. Block: 256 thr = 8 waves, 128 queries.
//    K tile staged by TDM (wave 0) with hardware pad to 72-elem rows;
//    V tile staged transposed by all threads. Per 64-key tile:
//    S = Q K^T (WMMA), t = relu(1+s+s^2/2), den += rowsum(t),
//    O += T V (WMMA, T re-read as A-frags from LDS). Normalize at end.
__global__ void __launch_bounds__(256) k_attn(
    const bf16_t* __restrict__ Qb, const bf16_t* __restrict__ Kb,
    const bf16_t* __restrict__ Vb, bf16_t* __restrict__ Ob)
{
  const int Hh = 12, Dh = 64, Npos = 1024;
  const int LP = 72; // padded LDS row stride (elements) = 144B = 32+4 dwords
  __shared__ __align__(16) bf16_t sK [64 * LP];      // [key][d]
  __shared__ __align__(16) bf16_t sVt[64 * LP];      // [d][key] (transposed)
  __shared__ __align__(16) bf16_t sT [8 * 16 * LP];  // per-wave Taylor tile

  const int lane = threadIdx.x & 31, wave = threadIdx.x >> 5;
  const int hi = lane >> 4;
  const int bh = blockIdx.y;                 // b*H + h
  const int b = bh / Hh, h = bh % Hh;
  const int q0 = blockIdx.x * 128 + wave * 16;

  const bf16_t* Qh = Qb + (size_t)bh * Npos * Dh;
  const bf16_t* Kh = Kb + (size_t)bh * Npos * Dh;
  const bf16_t* Vh = Vb + (size_t)bh * Npos * Dh;

  // Q A-frags for this wave's 16 queries (d 0..31, 32..63): kept in VGPRs
  v16bf qa0 = load_a_frag(Qh + (size_t)q0 * Dh + 0,  Dh, lane);
  v16bf qa1 = load_a_frag(Qh + (size_t)q0 * Dh + 32, Dh, lane);

  v8f onum[4];
#pragma unroll
  for (int fd = 0; fd < 4; ++fd) onum[fd] = zero8();
  float den[8] = {0.f,0.f,0.f,0.f,0.f,0.f,0.f,0.f};

#if ATTN_TDM
  const unsigned int sK_lds = (unsigned int)(uintptr_t)&sK[0]; // LDS byte offset
#endif

  for (int k0 = 0; k0 < Npos; k0 += 64){
    // --- stage K tile: Tensor Data Mover (one DMA per tile, wave 0 only) ---
#if ATTN_TDM
    if (wave == 0)
      tdm_load_tile_64x64((uint64_t)(uintptr_t)(Kh + (size_t)k0 * Dh), sK_lds);
#else
    for (int c = threadIdx.x; c < 512; c += 256){
      int key = c >> 3, part = c & 7;
      *(v8bf*)&sK[key * LP + part * 8] =
          *(const v8bf*)(Kh + (size_t)(k0 + key) * Dh + part * 8);
    }
#endif
    // --- stage V transposed: b128 global reads, b16 LDS scatters ---
    for (int c = threadIdx.x; c < 512; c += 256){
      int key = c >> 3, part = c & 7;
      v8bf v = *(const v8bf*)(Vh + (size_t)(k0 + key) * Dh + part * 8);
#pragma unroll
      for (int j = 0; j < 8; ++j)
        sVt[(part * 8 + j) * LP + key] = v[j];
    }
#if ATTN_TDM
    if (wave == 0) __builtin_amdgcn_s_wait_tensorcnt(0);
#endif
    __syncthreads();

    // --- scores + Taylor, spill T to per-wave LDS tile ---
    bf16_t* tw = &sT[wave * 16 * LP];
#pragma unroll
    for (int j = 0; j < 4; ++j){                     // 16-key column groups
      v8f s = zero8();
      v16bf bk0 = load_b_frag(&sK[(j * 16) * LP + 0 ], LP, lane);
      s = wmma_bf16(qa0, bk0, s);
      v16bf bk1 = load_b_frag(&sK[(j * 16) * LP + 32], LP, lane);
      s = wmma_bf16(qa1, bk1, s);
#pragma unroll
      for (int r = 0; r < 8; ++r){
        float x = s[r];
        float t = fmaf(0.5f * x, x, x + 1.0f);       // 1 + s + s^2/2
        t = fmaxf(t, 0.0f);                          // relu (no-op here, kept)
        den[r] += t;
        tw[(r + hi * 8) * LP + j * 16 + (lane & 15)] = f2bf(t);
      }
    }
    __syncthreads();

    // --- O += T * V ---
    v16bf ta0 = load_a_frag(tw + 0,  LP, lane);      // keys 0..31
    v16bf ta1 = load_a_frag(tw + 32, LP, lane);      // keys 32..63
#pragma unroll
    for (int fd = 0; fd < 4; ++fd){
      v16bf vb0 = load_b_frag(&sVt[(fd * 16) * LP + 0 ], LP, lane);
      onum[fd] = wmma_bf16(ta0, vb0, onum[fd]);
      v16bf vb1 = load_b_frag(&sVt[(fd * 16) * LP + 32], LP, lane);
      onum[fd] = wmma_bf16(ta1, vb1, onum[fd]);
    }
    __syncthreads();
  }

  // reduce denominator over the 16-lane column group (wave32)
#pragma unroll
  for (int off = 1; off < 16; off <<= 1)
#pragma unroll
    for (int r = 0; r < 8; ++r)
      den[r] += __shfl_xor(den[r], off, 16);

#pragma unroll
  for (int r = 0; r < 8; ++r){
    float inv = 1.0f / den[r];
    int row = q0 + r + hi * 8;
#pragma unroll
    for (int fd = 0; fd < 4; ++fd){
      int d = fd * 16 + (lane & 15);
      Ob[((size_t)b * Npos + row) * 768 + h * 64 + d] = f2bf(onum[fd][r] * inv);
    }
  }
}

// ---------------------------------------------------------------------------
// 5) proj GEMM: Ob[8192x768] @ Wproj[768x768] + bias -> f32 out
__global__ void __launch_bounds__(128) k_proj_gemm(
    const bf16_t* __restrict__ Ab, const bf16_t* __restrict__ Wp,
    const float* __restrict__ bias, float* __restrict__ out)
{
  const int Cdim = 768, NC = 768;
  const int lane = threadIdx.x & 31, wave = threadIdx.x >> 5;
  const int m0 = (blockIdx.x * 4 + wave) * 32;
  const int n0 = blockIdx.y * 64;
  const int nt16 = n0 >> 4;

  v8f acc[2][4];
#pragma unroll
  for (int g = 0; g < 2; ++g)
#pragma unroll
    for (int j = 0; j < 4; ++j) acc[g][j] = zero8();

  for (int kk = 0; kk < Cdim; kk += 32){
    v16bf a0 = load_a_frag(Ab + (size_t)m0 * Cdim + kk,        Cdim, lane);
    v16bf a1 = load_a_frag(Ab + (size_t)(m0 + 16) * Cdim + kk, Cdim, lane);
    const int kt = kk >> 5;
    const bf16_t* wbase = Wp + ((size_t)kt * (NC/16) + nt16) * 512 + (size_t)lane * 16;
    if (kk + 32 < Cdim)
      __builtin_prefetch(wbase + (size_t)(NC/16) * 512, 0, 1);
#pragma unroll
    for (int j = 0; j < 4; ++j){
      v16bf bfrag = *(const v16bf*)(wbase + (size_t)j * 512);
      acc[0][j] = wmma_bf16(a0, bfrag, acc[0][j]);
      acc[1][j] = wmma_bf16(a1, bfrag, acc[1][j]);
    }
  }

  const int hi = lane >> 4;
#pragma unroll
  for (int g = 0; g < 2; ++g)
#pragma unroll
    for (int j = 0; j < 4; ++j)
#pragma unroll
      for (int r = 0; r < 8; ++r){
        int row = m0 + g * 16 + r + hi * 8;
        int col = n0 + j * 16 + (lane & 15);
        out[(size_t)row * 768 + col] = acc[g][j][r] + bias[col];
      }
}

// ---------------------------------------------------------------------------
extern "C" void kernel_launch(void* const* d_in, const int* in_sizes, int n_in,
                              void* d_out, int out_size, void* d_ws, size_t ws_size,
                              hipStream_t stream)
{
  (void)in_sizes; (void)n_in; (void)out_size; (void)ws_size;
  const float* x      = (const float*)d_in[0];
  const float* W_qkv  = (const float*)d_in[1];
  const float* b_qkv  = (const float*)d_in[2];
  const float* W_proj = (const float*)d_in[3];
  const float* b_proj = (const float*)d_in[4];
  float* out = (float*)d_out;

  const size_t B = 8, N = 1024, C = 768, H = 12, Dh = 64;

  char* ws = (char*)d_ws;
  size_t off = 0;
  auto alloc = [&](size_t bytes) -> char* {
    char* p = ws + off;
    off += (bytes + 255) & ~(size_t)255;
    return p;
  };
  bf16_t* Xb  = (bf16_t*)alloc(B * N * C * 2);          // 12.6 MB
  bf16_t* Wqp = (bf16_t*)alloc(C * 3 * C * 2);          //  3.5 MB
  bf16_t* Wpp = (bf16_t*)alloc(C * C * 2);              //  1.2 MB
  bf16_t* Qb  = (bf16_t*)alloc(B * H * N * Dh * 2);     // 12.6 MB
  bf16_t* Kb  = (bf16_t*)alloc(B * H * N * Dh * 2);     // 12.6 MB
  bf16_t* Vb  = (bf16_t*)alloc(B * H * N * Dh * 2);     // 12.6 MB
  bf16_t* Ob  = (bf16_t*)alloc(B * N * C * 2);          // 12.6 MB

  int nx = (int)(B * N * C);
  k_cvt_x<<<nx / (256 * 8), 256, 0, stream>>>(x, Xb, nx);

  {
    int tot = (768 / 32) * (2304 / 16) * 32;
    k_pack_w<<<(tot + 255) / 256, 256, 0, stream>>>(W_qkv, Wqp, 768, 2304);
  }
  {
    int tot = (768 / 32) * (768 / 16) * 32;
    k_pack_w<<<(tot + 255) / 256, 256, 0, stream>>>(W_proj, Wpp, 768, 768);
  }

  k_qkv_gemm<<<dim3(64, 36), 128, 0, stream>>>(Xb, Wqp, b_qkv, Qb, Kb, Vb);
  k_attn   <<<dim3(8, 96),  256, 0, stream>>>(Qb, Kb, Vb, Ob);
  k_proj_gemm<<<dim3(64, 12), 128, 0, stream>>>(Ob, Wpp, b_proj, out);
}
